// RNN_36910948942238
// MI455X (gfx1250) — matrix operand order repaired and checked by reading, not compiled, not run on hardware
//
#include <hip/hip_runtime.h>
#include <hip/hip_bf16.h>

#define SEQ 32768
#define INP 512
#define HID 512
#define NG  1536   // 3*HID
#define NBLK 8     // workgroups cooperating on the sequential scan
#define WSTRIDE 514  // LDS row stride in bf16 elems; matches TDM pad (256 dw + 1 dw)

typedef __attribute__((ext_vector_type(16))) __bf16 v16bf;
typedef __attribute__((ext_vector_type(8)))  float  v8f;
typedef __attribute__((ext_vector_type(8)))  unsigned int v8u;
typedef __attribute__((ext_vector_type(4)))  unsigned int v4u;
typedef __attribute__((ext_vector_type(8)))  int v8i;
typedef __attribute__((ext_vector_type(4)))  int v4i;

__device__ __forceinline__ unsigned short f32_to_bf16_rne(float f) {
    unsigned int x = __float_as_uint(f);
    unsigned int r = (x + 0x7FFFu + ((x >> 16) & 1u)) >> 16;
    return (unsigned short)r;
}

// ---------------------------------------------------------------------------
// fp32 -> bf16 conversion (round to nearest even)
// ---------------------------------------------------------------------------
__global__ void cvt_bf16_kernel(const float* __restrict__ src,
                                unsigned short* __restrict__ dst, int n) {
    int i = blockIdx.x * blockDim.x + threadIdx.x;
    if (i < n) dst[i] = f32_to_bf16_rne(src[i]);
}

// ---------------------------------------------------------------------------
// zero the persistent-scan state (h carry + global barrier counter)
// ---------------------------------------------------------------------------
__global__ void init_state_kernel(float* __restrict__ hbuf,
                                  unsigned int* __restrict__ barcnt) {
    int t = threadIdx.x;
    if (t < HID) hbuf[t] = 0.0f;
    if (t + 256 < HID) hbuf[t + 256] = 0.0f;
    if (t == 0) *barcnt = 0u;
}

// ---------------------------------------------------------------------------
// igates = inputs @ W_ih^T + bias   via v_wmma_f32_16x16x32_bf16
// Register-blocked: each wave computes a 16x64 strip (4 N-tiles reuse one A
// fragment -> ~12.8 FLOP/B of load traffic). Block = 8 waves = 32x256 tile.
// A lane layout (16-bit A 16x32): row = lane%16, K-chunks {hi*8..+7, 16+hi*8..+7}
// B lane layout (16-bit B 32x16): col = lane%16, K-chunks {hi*8..+7, 16+hi*8..+7}
// C/D layout: VGPR r -> row m0 + (lane/16)*8 + r, col n + lane%16
// ---------------------------------------------------------------------------
__global__ __launch_bounds__(256) void igates_gemm_kernel(
    const unsigned short* __restrict__ A,    // SEQ x INP, bf16
    const unsigned short* __restrict__ W,    // NG  x INP, bf16 (weight_ih)
    const float* __restrict__ bias,          // NG
    float* __restrict__ C)                   // SEQ x NG, fp32
{
    const int lane = threadIdx.x & 31;
    const int wave = threadIdx.x >> 5;
    const int lo   = lane & 15;
    const int hi   = lane >> 4;

    const int m0 = blockIdx.x * 32 + (wave & 1) * 16;
    const int n0 = blockIdx.y * 256 + (wave >> 1) * 64;

    const unsigned short* arow = A + (size_t)(m0 + lo) * INP;
    const unsigned short* wrow0 = W + (size_t)(n0 +  0 + lo) * INP;
    const unsigned short* wrow1 = W + (size_t)(n0 + 16 + lo) * INP;
    const unsigned short* wrow2 = W + (size_t)(n0 + 32 + lo) * INP;
    const unsigned short* wrow3 = W + (size_t)(n0 + 48 + lo) * INP;

    v8f c0 = {}, c1 = {}, c2 = {}, c3 = {};
    for (int kb = 0; kb < INP; kb += 32) {
        if (kb + 32 < INP) __builtin_prefetch(arow + kb + 32, 0, 0);

        v4u a0 = *reinterpret_cast<const v4u*>(arow + kb + hi * 8);
        v4u a1 = *reinterpret_cast<const v4u*>(arow + kb + 16 + hi * 8);
        v8u au;
        au[0] = a0[0]; au[1] = a0[1]; au[2] = a0[2]; au[3] = a0[3];
        au[4] = a1[0]; au[5] = a1[1]; au[6] = a1[2]; au[7] = a1[3];
        v16bf av = __builtin_bit_cast(v16bf, au);

#define DO_TILE(WR, CC)                                                        \
        {                                                                      \
            v4u b0 = *reinterpret_cast<const v4u*>((WR) + kb + hi * 8);        \
            v4u b1 = *reinterpret_cast<const v4u*>((WR) + kb + 16 + hi * 8);   \
            v8u bu;                                                            \
            bu[0] = b0[0]; bu[1] = b0[1]; bu[2] = b0[2]; bu[3] = b0[3];        \
            bu[4] = b1[0]; bu[5] = b1[1]; bu[6] = b1[2]; bu[7] = b1[3];        \
            v16bf bv = __builtin_bit_cast(v16bf, bu);                          \
            CC = __builtin_amdgcn_wmma_f32_16x16x32_bf16(                      \
                     false, av, false, bv, (short)0, CC, false, false);        \
        }
        DO_TILE(wrow0, c0)
        DO_TILE(wrow1, c1)
        DO_TILE(wrow2, c2)
        DO_TILE(wrow3, c3)
#undef DO_TILE
    }

    const int mrow = m0 + hi * 8;
#define STORE_TILE(CC, J)                                                      \
    {                                                                          \
        const float bn = bias[n0 + (J) * 16 + lo];                             \
        float* crow = C + (size_t)mrow * NG + (n0 + (J) * 16 + lo);            \
        _Pragma("unroll")                                                      \
        for (int r = 0; r < 8; ++r) crow[(size_t)r * NG] = CC[r] + bn;         \
    }
    STORE_TILE(c0, 0)
    STORE_TILE(c1, 1)
    STORE_TILE(c2, 2)
    STORE_TILE(c3, 3)
#undef STORE_TILE
}

// ---------------------------------------------------------------------------
// Build a TDM descriptor + issue tensor_load_to_lds for one 64x512 bf16 chunk.
// Pad: after every 256 DWORDs (one row) insert 1 DWORD -> LDS row stride 514
// halves == WSTRIDE (bank-conflict-free for the 32-lane dot-product reads).
// This toolchain exposes the 6-arg builtin:
//   (uint32x4 g0, int32x8 g1, int32x4 g2, int32x4 g3, int32x8 g4, i32 cpol)
// ---------------------------------------------------------------------------
__device__ __forceinline__ void tdm_load_chunk(const unsigned short* gsrc,
                                               unsigned lds_byte_off) {
    unsigned long long gaddr = (unsigned long long)(uintptr_t)gsrc;

    v4u g0;
    g0[0] = 1u;                                           // count=1, user mode
    g0[1] = lds_byte_off;                                 // lds_addr (bytes)
    g0[2] = (unsigned)(gaddr & 0xFFFFFFFFu);              // global_addr[31:0]
    g0[3] = (unsigned)((gaddr >> 32) & 0x01FFFFFFu)       // global_addr[56:32]
          | (2u << 30);                                   // type = 2 ("image")

    v8i g1;
    g1[0] = (int)((1u << 16)      // data_size = 2 bytes
                | (1u << 20)      // pad_enable
                | (7u << 22)      // pad_interval: 256 DWORDs
                | (0u << 25));    // pad_amount: 1 DWORD
    g1[1] = (int)(512u << 16);    // tensor_dim0[15:0] = 512
    g1[2] = (int)(64u << 16);     // tensor_dim0[31:16]=0 | tensor_dim1[15:0]=64
    g1[3] = (int)(512u << 16);    // tensor_dim1[31:16]=0 | tile_dim0 = 512
    g1[4] = (int)64;              // tile_dim1 = 64, tile_dim2 = 0 (2D)
    g1[5] = (int)512;             // tensor_dim0_stride = 512 elems
    g1[6] = 0;                    // stride hi / tensor_dim1_stride lo
    g1[7] = 0;                    // tensor_dim1_stride hi

    v4i gz4 = {0, 0, 0, 0};                    // groups 2/3 unused (2D tile)
    v8i gz8 = {0, 0, 0, 0, 0, 0, 0, 0};        // trailing group, zero-filled

    __builtin_amdgcn_tensor_load_to_lds(g0, g1, gz4, gz4, gz8, 0);
}

// ---------------------------------------------------------------------------
// Sequential GRU scan: 8 persistent workgroups, each owns 64 hidden units and
// keeps its 192 rows of W_hh as bf16 in LDS (~197 KB of the WGP's 320 KB),
// staged by the Tensor Data Mover. Per step: read h (512 f32) -> 192
// LDS-resident dot-512 -> gate math on 64 units -> publish h slice -> device
// barrier (atomic; s_cluster_barrier is a NOP outside a cluster dispatch).
// ---------------------------------------------------------------------------
__global__ __launch_bounds__(256) void gru_scan_kernel(
    const unsigned short* __restrict__ WhhBf,  // NG x HID, bf16
    const float* __restrict__ igates,          // SEQ x NG, fp32
    const float* __restrict__ bias_n,          // HID
    float* __restrict__ hbuf,                  // HID carry
    unsigned int* __restrict__ barcnt,         // global barrier counter
    float* __restrict__ out)                   // SEQ x HID
{
    extern __shared__ unsigned char smem_raw[];
    unsigned short* Wl  = (unsigned short*)smem_raw;                 // 192 x WSTRIDE bf16
    float* h_s  = (float*)(smem_raw + (size_t)192 * WSTRIDE * 2);    // 512
    float* part = h_s + HID;                                         // 192

    const int tid = threadIdx.x;
    const int blk = blockIdx.x;          // 0..7
    const int ROWS = 192;                // 3 gates x 64 units

    // Stage this WG's W_hh rows via TDM: 3 chunks of 64 rows x 512 cols
    // (gate g, units blk*64..blk*64+63). Wave 0 issues the DMAs; TENSORcnt
    // orders them; workgroup barrier publishes LDS to the other waves.
    if (tid < 32) {
        const unsigned lds_base = (unsigned)__builtin_amdgcn_groupstaticsize();
        for (int g = 0; g < 3; ++g) {
            const unsigned short* gsrc =
                WhhBf + (size_t)(g * HID + blk * 64) * HID;
            unsigned lds_off = lds_base + (unsigned)(g * 64 * WSTRIDE * 2);
            tdm_load_chunk(gsrc, lds_off);
        }
        __builtin_amdgcn_s_wait_tensorcnt(0);
    }
    __syncthreads();

    unsigned int target = 0;

    for (int t = 0; t < SEQ; ++t) {
        // pull full h into LDS
        h_s[tid]       = hbuf[tid];
        h_s[tid + 256] = hbuf[tid + 256];
        __syncthreads();

        if (tid < ROWS) {
            const unsigned int* wr = (const unsigned int*)(Wl + tid * WSTRIDE);
            float acc = 0.0f;
#pragma unroll 8
            for (int k2 = 0; k2 < HID / 2; ++k2) {
                unsigned int p = wr[k2];                       // two packed bf16
                acc = fmaf(__uint_as_float(p << 16),          h_s[2 * k2],     acc);
                acc = fmaf(__uint_as_float(p & 0xFFFF0000u),  h_s[2 * k2 + 1], acc);
            }
            part[tid] = acc;
        }
        __syncthreads();

        if (tid < 64) {
            const int j = blk * 64 + tid;
            const float* ig = igates + (size_t)t * NG;
            float ir  = ig[j];
            float iz  = ig[HID + j];
            float inn = ig[2 * HID + j];
            float r = 1.0f / (1.0f + __expf(-(ir + part[tid])));
            float z = 1.0f / (1.0f + __expf(-(iz + part[64 + tid])));
            float n = tanhf(inn + r * (part[128 + tid] + bias_n[j]));
            float hnew = n + z * (h_s[j] - n);
            out[(size_t)t * HID + j] = hnew;
            hbuf[j] = hnew;
        }

        // device-wide barrier (monotonic counter; re-zeroed by init kernel)
        __threadfence();
        __syncthreads();
        if (tid == 0) {
            atomicAdd(barcnt, 1u);
            target += NBLK;
            while (__hip_atomic_load(barcnt, __ATOMIC_RELAXED,
                                     __HIP_MEMORY_SCOPE_AGENT) < target) {
                __builtin_amdgcn_s_sleep(1);
            }
        }
        __syncthreads();
        __threadfence();
        // NOP under plain dispatch; the real sync point under cluster dispatch
        __builtin_amdgcn_s_cluster_barrier();
    }
}

// ---------------------------------------------------------------------------
extern "C" void kernel_launch(void* const* d_in, const int* in_sizes, int n_in,
                              void* d_out, int out_size, void* d_ws, size_t ws_size,
                              hipStream_t stream) {
    (void)in_sizes; (void)n_in; (void)out_size; (void)ws_size;

    const float* inputs    = (const float*)d_in[0];  // SEQ x INP
    const float* weight_ih = (const float*)d_in[1];  // NG x INP
    const float* weight_hh = (const float*)d_in[2];  // NG x HID
    const float* bias      = (const float*)d_in[3];  // NG
    const float* bias_n    = (const float*)d_in[4];  // HID
    float* out = (float*)d_out;                      // SEQ x HID

    // workspace layout
    char* ws = (char*)d_ws;
    const size_t IGATES_B = (size_t)SEQ * NG * sizeof(float);     // 192 MB
    const size_t ABF_B    = (size_t)SEQ * INP * sizeof(unsigned short);
    const size_t WIH_B    = (size_t)NG * INP * sizeof(unsigned short);
    const size_t WHH_B    = (size_t)NG * HID * sizeof(unsigned short);
    float*          igates = (float*)ws;
    unsigned short* Abf    = (unsigned short*)(ws + IGATES_B);
    unsigned short* Wihbf  = (unsigned short*)(ws + IGATES_B + ABF_B);
    unsigned short* Whhbf  = (unsigned short*)(ws + IGATES_B + ABF_B + WIH_B);
    float*          hbuf   = (float*)(ws + IGATES_B + ABF_B + WIH_B + WHH_B);
    unsigned int*   barcnt = (unsigned int*)(hbuf + HID);

    // 1) convert inputs + both weights to bf16
    {
        int n = SEQ * INP;
        cvt_bf16_kernel<<<(n + 255) / 256, 256, 0, stream>>>(inputs, Abf, n);
    }
    {
        int n = NG * INP;
        cvt_bf16_kernel<<<(n + 255) / 256, 256, 0, stream>>>(weight_ih, Wihbf, n);
    }
    {
        int n = NG * HID;
        cvt_bf16_kernel<<<(n + 255) / 256, 256, 0, stream>>>(weight_hh, Whhbf, n);
    }

    // 2) reset scan state (h = 0, barrier counter = 0) every call
    init_state_kernel<<<1, 256, 0, stream>>>(hbuf, barcnt);

    // 3) igates GEMM: 1024 x 6 blocks, each 32x256 tile; 4 WMMAs per A fragment
    dim3 grid(SEQ / 32, NG / 256);
    igates_gemm_kernel<<<grid, 256, 0, stream>>>(Abf, Wihbf, bias, igates);

    // 4) sequential scan: 8 persistent WGs, ~200 KB dynamic LDS each
    size_t smem = (size_t)192 * WSTRIDE * sizeof(unsigned short)
                + (size_t)(HID + 192) * sizeof(float);
    (void)hipFuncSetAttribute((const void*)gru_scan_kernel,
                              hipFuncAttributeMaxDynamicSharedMemorySize,
                              (int)smem);
    gru_scan_kernel<<<NBLK, 256, smem, stream>>>(Whhbf, igates, bias_n,
                                                 hbuf, barcnt, out);
}
